// MyModel_22582938042951
// MI455X (gfx1250) — compile-verified, matrix operand order checked by reading, not run on hardware
//
#include <hip/hip_runtime.h>
#include <hip/hip_bf16.h>

typedef __attribute__((ext_vector_type(16))) __bf16 v16bf;
typedef __attribute__((ext_vector_type(8)))  float  v8f;

#define EMBED 768
#define ODIM  1024
#define HEADS 8
#define HDIM  128
#define KSTEPS (EMBED / 32)   // 24
#define NTILES (ODIM / 16)    // 64

// ---------------- init workspace ----------------
__global__ void init_ws(float* hout, float* denom, float* mmax, float* easum,
                        float* gsum, int N) {
  int i = blockIdx.x * blockDim.x + threadIdx.x;
  int stride = gridDim.x * blockDim.x;
  int totH = N * ODIM;
  for (int k = i; k < totH; k += stride) hout[k] = 0.0f;
  int totS = N * HEADS;
  for (int k = i; k < totS; k += stride) { denom[k] = 0.0f; mmax[k] = -__builtin_inff(); }
  for (int k = i; k < ODIM; k += stride) gsum[k] = 0.0f;
  if (i == 0) easum[0] = 0.0f;
}

// ---------------- mean of edge_attr ----------------
__global__ void reduce_ea(const float* ea, float* easum, int E) {
  __shared__ float s[256];
  int t = threadIdx.x;
  float p = 0.0f;
  for (int i = blockIdx.x * 256 + t; i < E; i += gridDim.x * 256) p += ea[i];
  s[t] = p; __syncthreads();
  for (int off = 128; off > 0; off >>= 1) { if (t < off) s[t] += s[t + off]; __syncthreads(); }
  if (t == 0) atomicAdd(easum, s[0]);
}

// ---------------- edge_coef[h] = sum_c W_edge[h*128+c]*att_edge[h*128+c] ----------------
__global__ void edge_coef_k(const float* W_edge, const float* att_edge, float* coef) {
  int h = threadIdx.x >> 5;
  int lane = threadIdx.x & 31;
  float p = 0.0f;
  for (int i = lane; i < HDIM; i += 32) p += W_edge[h * HDIM + i] * att_edge[h * HDIM + i];
  for (int off = 16; off > 0; off >>= 1) p += __shfl_down(p, off, 32);
  if (lane == 0) coef[h] = p;
}

// ---------------- pack W (768x1024 f32, row-major) into WMMA B-fragment bf16 layout --------
// layout: [nTile(64)][kStep(24)][lane(32)] x 16 bf16 (32B contiguous per lane)
// B 32x16 bf16 layout: lanes 0-15 col N=lane, K=kb..kb+7 & kb+16..kb+23 (kb=0);
//                      lanes 16-31 same with kb=8.
__global__ void pack_w(const float* W, __bf16* Wp) {
  int t = blockIdx.x * blockDim.x + threadIdx.x;   // < 64*24*32
  if (t >= NTILES * KSTEPS * 32) return;
  int lane  = t & 31;
  int kStep = (t >> 5) % KSTEPS;
  int nTile = t / (KSTEPS * 32);
  int n  = nTile * 16 + (lane & 15);
  int kb = kStep * 32 + ((lane & 16) ? 8 : 0);
  v16bf v;
  #pragma unroll
  for (int i = 0; i < 8; ++i) v[i]     = (__bf16)W[(size_t)(kb + i) * ODIM + n];
  #pragma unroll
  for (int i = 0; i < 8; ++i) v[8 + i] = (__bf16)W[(size_t)(kb + 16 + i) * ODIM + n];
  *(((v16bf*)Wp) + t) = v;
}

// ---------------- pack x (Nx768 f32, row-major) into WMMA A-fragment bf16 layout ----------
// layout: [mTile][kStep(24)][lane(32)] x 16 bf16 (32B contiguous per lane)
// A 16x32 bf16 layout: lanes 0-15 row M=lane, K=kb..kb+7 & kb+16..kb+23 (kb=0);
//                      lanes 16-31 same with kb=8.
__global__ void pack_x(const float* __restrict__ x, __bf16* __restrict__ Xp, int N) {
  int t = blockIdx.x * blockDim.x + threadIdx.x;   // Mtiles*24*32
  int Mtiles = (N + 15) >> 4;
  if (t >= Mtiles * KSTEPS * 32) return;
  int lane  = t & 31;
  int kStep = (t >> 5) % KSTEPS;
  int mTile = t / (KSTEPS * 32);
  int m = mTile * 16 + (lane & 15);
  if (m >= N) m = N - 1;                            // clamp (N multiple of 16 here)
  int kb = kStep * 32 + ((lane & 16) ? 8 : 0);
  const float* xp = x + (size_t)m * EMBED + kb;     // coalesced, K-contiguous
  float af[16];
  *(float4*)(af + 0)  = *(const float4*)(xp);
  *(float4*)(af + 4)  = *(const float4*)(xp + 4);
  *(float4*)(af + 8)  = *(const float4*)(xp + 16);
  *(float4*)(af + 12) = *(const float4*)(xp + 20);
  v16bf a;
  #pragma unroll
  for (int i = 0; i < 16; ++i) a[i] = (__bf16)af[i];
  *(((v16bf*)Xp) + t) = a;
}

// ---------------- GEMM: xs = x @ W via v_wmma_f32_16x16x32_bf16 ----------------
// one wave computes a 16(M) x 64(N) strip; 4 accumulators; K loop of 24 steps.
// A and B both pre-packed bf16 fragments: inner loop = 10x b128 loads + 4x WMMA.
__global__ void gemm_wmma(const __bf16* __restrict__ Xp, const __bf16* __restrict__ Wp,
                          float* __restrict__ xs, int N) {
  int gw   = blockIdx.x * (blockDim.x >> 5) + (threadIdx.x >> 5);
  int lane = threadIdx.x & 31;
  int Mtiles = (N + 15) >> 4;
  int strips = ODIM / 64;               // 16
  int tileM = gw / strips;
  int strip = gw % strips;
  if (tileM >= Mtiles) return;

  v8f acc0 = {}, acc1 = {}, acc2 = {}, acc3 = {};
  const v16bf* Xpv = (const v16bf*)Xp;
  const v16bf* Wpv = (const v16bf*)Wp;
  size_t abase = (size_t)tileM * KSTEPS * 32 + lane;
  size_t bbase = (size_t)(strip * 4) * KSTEPS * 32 + lane;

  for (int kStep = 0; kStep < KSTEPS; ++kStep) {
    v16bf a  = Xpv[abase + (size_t)kStep * 32];
    __builtin_prefetch((const __bf16*)(Xpv + abase + (size_t)(kStep + 1) * 32), 0, 3);
    v16bf b0 = Wpv[bbase + (size_t)kStep * 32];
    v16bf b1 = Wpv[bbase + (size_t)(KSTEPS + kStep) * 32];
    v16bf b2 = Wpv[bbase + (size_t)(2 * KSTEPS + kStep) * 32];
    v16bf b3 = Wpv[bbase + (size_t)(3 * KSTEPS + kStep) * 32];
    acc0 = __builtin_amdgcn_wmma_f32_16x16x32_bf16(false, a, false, b0, (short)0, acc0, false, false);
    acc1 = __builtin_amdgcn_wmma_f32_16x16x32_bf16(false, a, false, b1, (short)0, acc1, false, false);
    acc2 = __builtin_amdgcn_wmma_f32_16x16x32_bf16(false, a, false, b2, (short)0, acc2, false, false);
    acc3 = __builtin_amdgcn_wmma_f32_16x16x32_bf16(false, a, false, b3, (short)0, acc3, false, false);
  }

  // C/D layout: lanes 0-15 N=lane (M rows 0-7 in vgprs 0-7); lanes 16-31 rows 8-15
  int nc    = lane & 15;
  int mbase = tileM * 16 + ((lane & 16) ? 8 : 0);
  v8f accs[4] = {acc0, acc1, acc2, acc3};
  #pragma unroll
  for (int tIdx = 0; tIdx < 4; ++tIdx) {
    int n = (strip * 4 + tIdx) * 16 + nc;
    #pragma unroll
    for (int r = 0; r < 8; ++r) {
      int mm = mbase + r;
      if (mm < N) xs[(size_t)mm * ODIM + n] = accs[tIdx][r];
    }
  }
}

// ---------------- per-node attention logits ----------------
__global__ void node_logits(const float* __restrict__ xs, const float* __restrict__ att_src,
                            const float* __restrict__ att_dst, float* asrc, float* adst, int N) {
  int n = blockIdx.x;
  int h = threadIdx.x >> 5;
  int lane = threadIdx.x & 31;
  const float* row = xs + (size_t)n * ODIM + h * HDIM;
  float4 v = *(const float4*)(row + lane * 4);
  float4 s = *(const float4*)(att_src + h * HDIM + lane * 4);
  float4 d = *(const float4*)(att_dst + h * HDIM + lane * 4);
  float ps = v.x * s.x + v.y * s.y + v.z * s.z + v.w * s.w;
  float pd = v.x * d.x + v.y * d.y + v.z * d.z + v.w * d.w;
  for (int off = 16; off > 0; off >>= 1) {
    ps += __shfl_down(ps, off, 32);
    pd += __shfl_down(pd, off, 32);
  }
  if (lane == 0) { asrc[n * HEADS + h] = ps; adst[n * HEADS + h] = pd; }
}

// ---------------- float atomic max via int ordering trick ----------------
__device__ __forceinline__ void atomicMaxF(float* addr, float v) {
  if (v >= 0.0f) atomicMax((int*)addr, __float_as_int(v));
  else           atomicMin((unsigned int*)addr, __float_as_uint(v));
}

// ---------------- edge alpha + segment max ----------------
__global__ void edge_alpha(const int* __restrict__ ei, const float* __restrict__ ea,
                           const float* __restrict__ easum, const float* __restrict__ asrc,
                           const float* __restrict__ adst, const float* __restrict__ coef,
                           float* alpha, float* mmax, int E, int N) {
  int e = blockIdx.x * blockDim.x + threadIdx.x;
  if (e >= E + N) return;
  int s, d; float eattr;
  if (e < E) { s = ei[e]; d = ei[E + e]; eattr = ea[e]; }
  else       { s = d = e - E; eattr = easum[0] / (float)E; }
  #pragma unroll
  for (int h = 0; h < HEADS; ++h) {
    float v = asrc[s * HEADS + h] + adst[d * HEADS + h] + eattr * coef[h];
    v = (v > 0.0f) ? v : 0.2f * v;            // leaky_relu(0.2)
    alpha[(size_t)e * HEADS + h] = v;
    atomicMaxF(&mmax[d * HEADS + h], v);
  }
}

// ---------------- exp + segment sum ----------------
__global__ void edge_exp(const int* __restrict__ ei, float* alpha, const float* __restrict__ mmax,
                         float* denom, int E, int N) {
  int e = blockIdx.x * blockDim.x + threadIdx.x;
  if (e >= E + N) return;
  int d = (e < E) ? ei[E + e] : (e - E);
  #pragma unroll
  for (int h = 0; h < HEADS; ++h) {
    float x = __expf(alpha[(size_t)e * HEADS + h] - mmax[d * HEADS + h]);
    alpha[(size_t)e * HEADS + h] = x;
    atomicAdd(&denom[d * HEADS + h], x);
  }
}

// ---------------- weighted message scatter: h[dst] += xs[src] * w ----------------
__global__ void scatter_msg(const int* __restrict__ ei, const float* __restrict__ ex,
                            const float* __restrict__ denom, const float* __restrict__ xs,
                            float* hout, int E, int N) {
  int e = blockIdx.x;
  int t = threadIdx.x;                  // 256 threads x 4 channels = 1024
  int s, d;
  if (e < E) { s = ei[e]; d = ei[E + e]; } else { s = d = e - E; }
  int c = t * 4;
  int h = c >> 7;                        // all 4 channels in the same head
  float w = ex[(size_t)e * HEADS + h] / denom[d * HEADS + h];
  float4 v = *(const float4*)(xs + (size_t)s * ODIM + c);
  float* o = hout + (size_t)d * ODIM + c;
  atomicAdd(o + 0, v.x * w);
  atomicAdd(o + 1, v.y * w);
  atomicAdd(o + 2, v.z * w);
  atomicAdd(o + 3, v.w * w);
}

// ---------------- column sums of h for global mean pool ----------------
__global__ void colsum(const float* __restrict__ hout, float* gsum, int N) {
  int c = blockIdx.x * blockDim.x + threadIdx.x;
  if (c >= ODIM) return;
  float s = 0.0f;
  for (int n = 0; n < N; ++n) s += hout[(size_t)n * ODIM + c];
  gsum[c] = s;
}

// ---------------- fused concat + classifier ----------------
__global__ void final_logits(const float* __restrict__ gsum, const float* __restrict__ hout,
                             const float* __restrict__ bias, const float* __restrict__ clf_W,
                             const float* __restrict__ clf_b, const int* text_idx,
                             const int* image_idx, float* out, int N) {
  __shared__ float s0[256], s1[256];
  int t = threadIdx.x;
  int ti = *text_idx, im = *image_idx;
  float p0 = 0.0f, p1 = 0.0f;
  for (int i = t; i < 3 * ODIM; i += 256) {
    int c = i & (ODIM - 1);
    float f;
    if (i < ODIM)           f = gsum[c] / (float)N + bias[c];
    else if (i < 2 * ODIM)  f = hout[(size_t)ti * ODIM + c] + bias[c];
    else                    f = hout[(size_t)im * ODIM + c] + bias[c];
    p0 += f * clf_W[i * 2 + 0];
    p1 += f * clf_W[i * 2 + 1];
  }
  s0[t] = p0; s1[t] = p1; __syncthreads();
  for (int off = 128; off > 0; off >>= 1) {
    if (t < off) { s0[t] += s0[t + off]; s1[t] += s1[t + off]; }
    __syncthreads();
  }
  if (t == 0) { out[0] = s0[0] + clf_b[0]; out[1] = s1[0] + clf_b[1]; }
}

extern "C" void kernel_launch(void* const* d_in, const int* in_sizes, int n_in,
                              void* d_out, int out_size, void* d_ws, size_t ws_size,
                              hipStream_t stream) {
  const float* x         = (const float*)d_in[0];
  const int*   ei        = (const int*)d_in[1];
  const float* ea        = (const float*)d_in[2];
  const int*   text_idx  = (const int*)d_in[3];
  const int*   image_idx = (const int*)d_in[4];
  const float* W         = (const float*)d_in[5];
  const float* att_src   = (const float*)d_in[6];
  const float* att_dst   = (const float*)d_in[7];
  const float* W_edge    = (const float*)d_in[8];
  const float* att_edge  = (const float*)d_in[9];
  const float* bias      = (const float*)d_in[10];
  const float* clf_W     = (const float*)d_in[11];
  const float* clf_b     = (const float*)d_in[12];
  float* out = (float*)d_out;

  int N = in_sizes[0] / EMBED;         // 10000
  int E = in_sizes[2];                 // 160000
  int Etot = E + N;
  int Mtiles = (N + 15) / 16;

  char* ws = (char*)d_ws;
  size_t off = 0;
  float*  xs    = (float*)(ws + off);  off += (size_t)N * ODIM * 4;
  float*  hout  = (float*)(ws + off);  off += (size_t)N * ODIM * 4;
  __bf16* Wp    = (__bf16*)(ws + off); off += (size_t)EMBED * ODIM * 2;
  __bf16* Xp    = (__bf16*)(ws + off); off += (size_t)Mtiles * 16 * EMBED * 2;
  float*  asrc  = (float*)(ws + off);  off += (size_t)N * HEADS * 4;
  float*  adst  = (float*)(ws + off);  off += (size_t)N * HEADS * 4;
  float*  alpha = (float*)(ws + off);  off += (size_t)Etot * HEADS * 4;
  float*  mmax  = (float*)(ws + off);  off += (size_t)N * HEADS * 4;
  float*  denom = (float*)(ws + off);  off += (size_t)N * HEADS * 4;
  float*  gsum  = (float*)(ws + off);  off += ODIM * 4;
  float*  coef  = (float*)(ws + off);  off += 64;
  float*  easum = (float*)(ws + off);  off += 64;

  init_ws<<<512, 256, 0, stream>>>(hout, denom, mmax, easum, gsum, N);
  reduce_ea<<<64, 256, 0, stream>>>(ea, easum, E);
  edge_coef_k<<<1, 256, 0, stream>>>(W_edge, att_edge, coef);
  pack_w<<<(NTILES * KSTEPS * 32 + 255) / 256, 256, 0, stream>>>(W, Wp);
  pack_x<<<(Mtiles * KSTEPS * 32 + 255) / 256, 256, 0, stream>>>(x, Xp, N);

  int waves = Mtiles * (ODIM / 64);
  gemm_wmma<<<(waves + 7) / 8, 256, 0, stream>>>(Xp, Wp, xs, N);

  node_logits<<<N, 256, 0, stream>>>(xs, att_src, att_dst, asrc, adst, N);
  edge_alpha<<<(Etot + 255) / 256, 256, 0, stream>>>(ei, ea, easum, asrc, adst, coef,
                                                     alpha, mmax, E, N);
  edge_exp<<<(Etot + 255) / 256, 256, 0, stream>>>(ei, alpha, mmax, denom, E, N);
  scatter_msg<<<Etot, 256, 0, stream>>>(ei, alpha, denom, xs, hout, E, N);
  colsum<<<(ODIM + 255) / 256, 256, 0, stream>>>(hout, gsum, N);
  final_logits<<<1, 256, 0, stream>>>(gsum, hout, bias, clf_W, clf_b,
                                      text_idx, image_idx, out, N);
}
